// TrajPreAttnAvgLongUser_38809324487098
// MI455X (gfx1250) — matrix operand order repaired and checked by reading, not compiled, not run on hardware
//
#include <hip/hip_runtime.h>

// ---------------------------------------------------------------------------
// Types for CDNA5 WMMA (wave32, 16x16x32 bf16 -> f32)
// ---------------------------------------------------------------------------
typedef __bf16 bf16t;
typedef __attribute__((ext_vector_type(16))) __bf16 v16bf;
typedef __attribute__((ext_vector_type(8)))  __bf16 v8bf;
typedef __attribute__((ext_vector_type(8)))  float  v8f;
typedef __attribute__((ext_vector_type(4)))  float  v4f;

// ---------------------------------------------------------------------------
// WMMA fragment loaders (layouts per CDNA5 ISA 7.12.2)
// A: 16x32 bf16, row-major source (M rows, K contiguous)
//   lane<16 : M=lane,    elems 0..7 -> K=k0..k0+7,   elems 8..15 -> K=k0+16..k0+23
//   lane>=16: M=lane-16, elems 0..7 -> K=k0+8..k0+15,elems 8..15 -> K=k0+24..k0+31
// B: 32x16 bf16 from "NT" source (rows are N, K contiguous)
//   lane&15 = N, K order same as A (elems 0..7 -> k0+8*(lane>>4), 8..15 -> k0+16+8*(lane>>4))
// ---------------------------------------------------------------------------
__device__ inline v16bf load_a_frag(const bf16t* __restrict__ A, int lda,
                                    int row, int k0, int lane) {
  const int m   = row + (lane & 15);
  const int sel = lane >> 4;
  const bf16t* p = A + (size_t)m * lda + k0;
  v8bf lo = *(const v8bf*)(p + sel * 8);
  v8bf hi = *(const v8bf*)(p + 16 + sel * 8);
  v16bf r;
#pragma unroll
  for (int i = 0; i < 8; ++i) { r[i] = lo[i]; r[i + 8] = hi[i]; }
  return r;
}

__device__ inline v16bf load_b_frag_bf16(const bf16t* __restrict__ B, int ldb,
                                         int row, int k0, int lane) {
  const int n = row + (lane & 15);
  const int k = k0 + (lane >> 4) * 8;
  const bf16t* p = B + (size_t)n * ldb + k;
  v8bf lo = *(const v8bf*)p;
  v8bf hi = *(const v8bf*)(p + 16);
  v16bf r;
#pragma unroll
  for (int i = 0; i < 8; ++i) { r[i] = lo[i]; r[i + 8] = hi[i]; }
  return r;
}

__device__ inline v16bf load_b_frag_f32(const float* __restrict__ B, int ldb,
                                        int row, int k0, int Nrows, int lane) {
  const int n = row + (lane & 15);
  const int k = k0 + (lane >> 4) * 8;
  v16bf r;
  if (n < Nrows) {
    const float* p = B + (size_t)n * ldb + k;
    v8f f0 = *(const v8f*)p;
    v8f f1 = *(const v8f*)(p + 16);
    if (n + 1 < Nrows) __builtin_prefetch(p + 32, 0, 1);  // next K-chunk
#pragma unroll
    for (int i = 0; i < 8; ++i) { r[i] = (bf16t)f0[i]; r[i + 8] = (bf16t)f1[i]; }
  } else {
#pragma unroll
    for (int i = 0; i < 16; ++i) r[i] = (bf16t)0.f;
  }
  return r;
}

// ---------------------------------------------------------------------------
// Generic NT GEMM: C[m][n] = sum_k A[m][k] * B[n][k]  (A bf16; B bf16 or fp32)
// Block = 128 threads (4 waves), tile 128(M)x128(N); wave tile 64x64
// (16 x v_wmma_f32_16x16x32_bf16 per 32-wide K step, 4x A/B fragment reuse).
// blockIdx.x = M-block (fastest) so concurrent M-blocks share the streamed
// B tile in L2 (critical for the 435MB Wf pass).
// Epilogue: optional bias, tanh, f32 / bf16 / transposed-bf16 stores.
// ---------------------------------------------------------------------------
template <int BIAS, int TANH, int STF32, int STB16, int STB16T, int BFP32>
__global__ __launch_bounds__(128) void gemm_nt(
    const bf16t* __restrict__ A, int lda,
    const void* __restrict__ Bmat, int ldb,
    const float* __restrict__ bias,
    float* __restrict__ C, int ldc,
    bf16t* __restrict__ Cb, int ldcb,
    bf16t* __restrict__ CbT, int ldct,
    int Nn, int K) {
  const int lane  = threadIdx.x & 31;
  const int wid   = threadIdx.x >> 5;
  const int mbase = blockIdx.x * 128 + (wid >> 1) * 64;
  const int nbase = blockIdx.y * 128 + (wid & 1) * 64;

  v8f acc[4][4];
#pragma unroll
  for (int i = 0; i < 4; ++i)
#pragma unroll
    for (int j = 0; j < 4; ++j)
#pragma unroll
      for (int e = 0; e < 8; ++e) acc[i][j][e] = 0.f;

  for (int k0 = 0; k0 < K; k0 += 32) {
    v16bf afr[4], bfr[4];
#pragma unroll
    for (int t = 0; t < 4; ++t)
      afr[t] = load_a_frag(A, lda, mbase + t * 16, k0, lane);
#pragma unroll
    for (int t = 0; t < 4; ++t) {
      if (BFP32)
        bfr[t] = load_b_frag_f32((const float*)Bmat, ldb, nbase + t * 16, k0, Nn, lane);
      else
        bfr[t] = load_b_frag_bf16((const bf16t*)Bmat, ldb, nbase + t * 16, k0, lane);
    }
#pragma unroll
    for (int i = 0; i < 4; ++i)
#pragma unroll
      for (int j = 0; j < 4; ++j)
        acc[i][j] = __builtin_amdgcn_wmma_f32_16x16x32_bf16(
            false, afr[i], false, bfr[j], (short)0, acc[i][j], false, false);
  }

  // C/D layout: elem v -> M = v + (lane>=16 ? 8 : 0), N = lane & 15
  const int mloc = (lane >> 4) * 8;
  const int nloc = lane & 15;
#pragma unroll
  for (int j = 0; j < 4; ++j) {
    const int n = nbase + j * 16 + nloc;
    if (BFP32 && n >= Nn) continue;  // ragged N only on the fp32-B path
    const float bv = BIAS ? bias[n] : 0.f;
#pragma unroll
    for (int i = 0; i < 4; ++i) {
#pragma unroll
      for (int e = 0; e < 8; ++e) {
        const int m = mbase + i * 16 + mloc + e;
        float v = acc[i][j][e] + bv;
        if (TANH) v = tanhf(v);
        if (STF32)  C[(size_t)m * ldc + n] = v;
        if (STB16)  Cb[(size_t)m * ldcb + n] = (bf16t)v;
        if (STB16T) CbT[(size_t)n * ldct + m] = (bf16t)v;
      }
    }
  }
}

// ---------------------------------------------------------------------------
// Exclusive prefix-sum of history_count -> offsets (single block)
// ---------------------------------------------------------------------------
__global__ __launch_bounds__(1024) void scan_offsets(const int* __restrict__ counts,
                                                     int* __restrict__ offs, int NH) {
  __shared__ int part[1024];
  const int tid = threadIdx.x;
  const int chunk = (NH + 1023) >> 10;
  int s = 0;
  for (int c = 0; c < chunk; ++c) {
    const int i = tid * chunk + c;
    if (i < NH) s += counts[i];
  }
  part[tid] = s;
  __syncthreads();
  for (int off = 1; off < 1024; off <<= 1) {
    const int add = (tid >= off) ? part[tid - off] : 0;
    __syncthreads();
    part[tid] += add;
    __syncthreads();
  }
  int run = (tid == 0) ? 0 : part[tid - 1];
  for (int c = 0; c < chunk; ++c) {
    const int i = tid * chunk + c;
    if (i < NH) { offs[i] = run; run += counts[i]; }
  }
}

// history feature rows: [mean(emb_loc[seg]) (256) | emb_tim[first] (32)] -> bf16
__global__ __launch_bounds__(64) void build_feat(
    const int* __restrict__ hloc, const int* __restrict__ htim,
    const int* __restrict__ counts, const int* __restrict__ offs,
    const float* __restrict__ embL, const float* __restrict__ embT,
    bf16t* __restrict__ feat) {
  const int i = blockIdx.x;
  const int cnt = counts[i];
  const int off = offs[i];
  const float inv = 1.f / (float)(cnt > 0 ? cnt : 1);
  for (int d = threadIdx.x; d < 288; d += 64) {
    float v;
    if (d < 256) {
      float s = 0.f;
      for (int c = 0; c < cnt; ++c)
        s += embL[(size_t)hloc[off + c] * 256 + d];
      v = s * inv;
    } else {
      v = embT[(size_t)htim[off] * 32 + (d - 256)];
    }
    feat[(size_t)i * 288 + d] = (bf16t)v;
  }
}

// x[t] = [emb_loc[loc[t]] | emb_tim[tim[t]]] -> bf16 (SEQ x 288)
__global__ void build_x(const int* __restrict__ loc, const int* __restrict__ tim,
                        const float* __restrict__ embL, const float* __restrict__ embT,
                        bf16t* __restrict__ x, int SEQ) {
  const int idx = blockIdx.x * blockDim.x + threadIdx.x;
  if (idx >= SEQ * 288) return;
  const int t = idx / 288, d = idx - t * 288;
  const float v = (d < 256) ? embL[(size_t)loc[t] * 256 + d]
                            : embT[(size_t)tim[t] * 32 + (d - 256)];
  x[idx] = (bf16t)v;
}

__global__ void f32_to_bf16(const float* __restrict__ src, bf16t* __restrict__ dst, long n) {
  const long i = (long)blockIdx.x * blockDim.x + threadIdx.x;
  if (i < n) dst[i] = (bf16t)src[i];
}

// ---------------------------------------------------------------------------
// Two GRU directions as two persistent workgroups (grid=2, concurrent).
// h (256) and gh (768) live in LDS; Whh (768x256 fp32, 786KB) streamed from L2
// every step. Writes hidden state for t >= SEQ-TGT directly as bf16 into the
// fused [q|context|uid] operand (cols 0..255 fwd, 256..511 bwd).
// ---------------------------------------------------------------------------
__global__ __launch_bounds__(768) void gru2_kernel(
    const float* __restrict__ gi_f, const float* __restrict__ gi_b,
    const float* __restrict__ Whh_f, const float* __restrict__ Whh_b,
    const float* __restrict__ bhh_f, const float* __restrict__ bhh_b,
    bf16t* __restrict__ outq, int SEQ, int TGT) {
  const int dir = blockIdx.x;
  const float* gi  = dir ? gi_b  : gi_f;
  const float* Whh = dir ? Whh_b : Whh_f;
  const float* bhh = dir ? bhh_b : bhh_f;
  const int col0 = dir ? 256 : 0;

  __shared__ float h[256];
  __shared__ float gh[768];
  const int tid = threadIdx.x;
  if (tid < 256) h[tid] = 0.f;
  const float bj = bhh[tid];
  const v4f* wrow = (const v4f*)(Whh + (size_t)tid * 256);
  __syncthreads();

  for (int s = 0; s < SEQ; ++s) {
    const int t = dir ? (SEQ - 1 - s) : s;
    // gh = Whh @ h + bhh   (768 dots of length 256)
    float acc = bj;
#pragma unroll 8
    for (int k4 = 0; k4 < 64; ++k4) {
      const v4f w = wrow[k4];
      acc += w[0] * h[k4 * 4] + w[1] * h[k4 * 4 + 1] +
             w[2] * h[k4 * 4 + 2] + w[3] * h[k4 * 4 + 3];
    }
    gh[tid] = acc;
    __syncthreads();
    if (tid < 256) {
      const float* gir = gi + (size_t)t * 768;
      const float r  = 1.f / (1.f + expf(-(gir[tid] + gh[tid])));
      const float z  = 1.f / (1.f + expf(-(gir[256 + tid] + gh[256 + tid])));
      const float nn = tanhf(gir[512 + tid] + r * gh[512 + tid]);
      const float hn = (1.f - z) * nn + z * h[tid];
      h[tid] = hn;
      const int m = t - (SEQ - TGT);
      if (m >= 0) outq[(size_t)m * 1088 + col0 + tid] = (bf16t)hn;
    }
    __syncthreads();
  }
}

// out[:,1024:1088] = emb_uid[uid] broadcast
__global__ void fill_uid(const float* __restrict__ embU, const int* __restrict__ uid,
                         bf16t* __restrict__ outq, int TGT) {
  const int idx = blockIdx.x * blockDim.x + threadIdx.x;
  if (idx >= TGT * 64) return;
  const int m = idx >> 6, d = idx & 63;
  outq[(size_t)m * 1088 + 1024 + d] = (bf16t)embU[(size_t)uid[0] * 64 + d];
}

// one block per attention row: softmax(scores) -> bf16 probabilities
__global__ __launch_bounds__(256) void softmax_rows(const float* __restrict__ S,
                                                    bf16t* __restrict__ P, int NH) {
  __shared__ float red[256];
  const int tid = threadIdx.x;
  const float* srow = S + (size_t)blockIdx.x * NH;
  float mx = -3.402823e38f;
  for (int i = tid; i < NH; i += 256) mx = fmaxf(mx, srow[i]);
  red[tid] = mx; __syncthreads();
  for (int s = 128; s > 0; s >>= 1) { if (tid < s) red[tid] = fmaxf(red[tid], red[tid + s]); __syncthreads(); }
  mx = red[0]; __syncthreads();
  float sum = 0.f;
  for (int i = tid; i < NH; i += 256) sum += expf(srow[i] - mx);
  red[tid] = sum; __syncthreads();
  for (int s = 128; s > 0; s >>= 1) { if (tid < s) red[tid] += red[tid + s]; __syncthreads(); }
  const float inv = 1.f / red[0];
  bf16t* prow = P + (size_t)blockIdx.x * NH;
  for (int i = tid; i < NH; i += 256) prow[i] = (bf16t)(expf(srow[i] - mx) * inv);
}

// per-row logsumexp over the 100000-wide logits
__global__ __launch_bounds__(512) void row_lse(const float* __restrict__ X,
                                               float* __restrict__ lse, int N) {
  __shared__ float red[512];
  const int tid = threadIdx.x;
  const float* x = X + (size_t)blockIdx.x * N;
  float mx = -3.402823e38f;
  for (int i = tid; i < N; i += 512) mx = fmaxf(mx, x[i]);
  red[tid] = mx; __syncthreads();
  for (int s = 256; s > 0; s >>= 1) { if (tid < s) red[tid] = fmaxf(red[tid], red[tid + s]); __syncthreads(); }
  mx = red[0]; __syncthreads();
  float sum = 0.f;
  for (int i = tid; i < N; i += 512) sum += expf(x[i] - mx);
  red[tid] = sum; __syncthreads();
  for (int s = 256; s > 0; s >>= 1) { if (tid < s) red[tid] += red[tid + s]; __syncthreads(); }
  if (tid == 0) lse[blockIdx.x] = mx + logf(red[0]);
}

__global__ void log_sub_vec(float* __restrict__ X, const float* __restrict__ lse,
                            int N, long n4) {
  const long i = (long)blockIdx.x * blockDim.x + threadIdx.x;
  if (i >= n4) return;
  float4* X4 = (float4*)X;
  float4 v = X4[i];
  const float l = lse[(long)(i * 4) / (long)N];
  v.x -= l; v.y -= l; v.z -= l; v.w -= l;
  X4[i] = v;
}

__global__ void log_sub_sc(float* __restrict__ X, const float* __restrict__ lse,
                           int N, long total) {
  const long i = (long)blockIdx.x * blockDim.x + threadIdx.x;
  if (i < total) X[i] -= lse[i / (long)N];
}

// ---------------------------------------------------------------------------
// Host orchestration (graph-capture safe: only kernel launches on `stream`)
// ---------------------------------------------------------------------------
extern "C" void kernel_launch(void* const* d_in, const int* in_sizes, int n_in,
                              void* d_out, int out_size, void* d_ws, size_t ws_size,
                              hipStream_t stream) {
  (void)n_in; (void)ws_size;
  const int*   loc     = (const int*)d_in[0];
  const int*   tim     = (const int*)d_in[1];
  const int*   hloc    = (const int*)d_in[2];
  const int*   htim    = (const int*)d_in[3];
  const int*   hcnt    = (const int*)d_in[4];
  const int*   uid     = (const int*)d_in[5];
  // d_in[6] = target_len (device scalar); derived from out_size instead.
  const float* embL    = (const float*)d_in[7];
  const float* embT    = (const float*)d_in[8];
  const float* embU    = (const float*)d_in[9];
  const float* W_attn  = (const float*)d_in[10];
  const float* b_attn  = (const float*)d_in[11];
  const float* Wih_f   = (const float*)d_in[12];
  const float* Whh_f   = (const float*)d_in[13];
  const float* bih_f   = (const float*)d_in[14];
  const float* bhh_f   = (const float*)d_in[15];
  const float* Wih_b   = (const float*)d_in[16];
  const float* Whh_b   = (const float*)d_in[17];
  const float* bih_b   = (const float*)d_in[18];
  const float* bhh_b   = (const float*)d_in[19];
  const float* Wf      = (const float*)d_in[20];
  const float* bf_bias = (const float*)d_in[21];

  const int SEQ = in_sizes[0];      // 2048
  const int NH  = in_sizes[4];      // 4096
  const int LOC = in_sizes[21];     // 100000
  const int TGT = out_size / LOC;   // 1024
  const int IN = 288, G = 768, H2 = 512, KO = 1088;

  char* p = (char*)d_ws;
  auto carve = [&](size_t bytes) -> void* {
    void* r = (void*)p;
    p += (bytes + 255) & ~(size_t)255;
    return r;
  };
  int*   offs    = (int*)  carve((size_t)NH * 4);
  bf16t* feat_b  = (bf16t*)carve((size_t)NH * IN * 2);
  bf16t* Wattn_b = (bf16t*)carve((size_t)H2 * IN * 2);
  bf16t* hist_b  = (bf16t*)carve((size_t)NH * H2 * 2);   // (NH x 512)
  bf16t* histT_b = (bf16t*)carve((size_t)H2 * NH * 2);   // (512 x NH)
  bf16t* x_b     = (bf16t*)carve((size_t)SEQ * IN * 2);
  bf16t* Wihf_b  = (bf16t*)carve((size_t)G * IN * 2);
  bf16t* Wihb_b  = (bf16t*)carve((size_t)G * IN * 2);
  float* gif     = (float*)carve((size_t)SEQ * G * 4);
  float* gib     = (float*)carve((size_t)SEQ * G * 4);
  bf16t* out_b   = (bf16t*)carve((size_t)TGT * KO * 2);  // fused [q|ctx|uid]
  float* scores  = (float*)carve((size_t)TGT * NH * 4);
  bf16t* attn_b  = (bf16t*)carve((size_t)TGT * NH * 2);
  float* lse     = (float*)carve((size_t)TGT * 4);

  // --- history branch ---
  scan_offsets<<<1, 1024, 0, stream>>>(hcnt, offs, NH);
  build_feat<<<NH, 64, 0, stream>>>(hloc, htim, hcnt, offs, embL, embT, feat_b);
  { long n = (long)H2 * IN; f32_to_bf16<<<(n + 255) / 256, 256, 0, stream>>>(W_attn, Wattn_b, n); }
  gemm_nt<1, 1, 0, 1, 1, 0><<<dim3(NH / 128, H2 / 128), 128, 0, stream>>>(
      feat_b, IN, Wattn_b, IN, b_attn, nullptr, 0, hist_b, H2, histT_b, NH, H2, IN);

  // --- GRU input projections ---
  { long n = (long)SEQ * IN; build_x<<<(n + 255) / 256, 256, 0, stream>>>(loc, tim, embL, embT, x_b, SEQ); }
  { long n = (long)G * IN;
    f32_to_bf16<<<(n + 255) / 256, 256, 0, stream>>>(Wih_f, Wihf_b, n);
    f32_to_bf16<<<(n + 255) / 256, 256, 0, stream>>>(Wih_b, Wihb_b, n); }
  gemm_nt<1, 0, 1, 0, 0, 0><<<dim3(SEQ / 128, G / 128), 128, 0, stream>>>(
      x_b, IN, Wihf_b, IN, bih_f, gif, G, nullptr, 0, nullptr, 0, G, IN);
  gemm_nt<1, 0, 1, 0, 0, 0><<<dim3(SEQ / 128, G / 128), 128, 0, stream>>>(
      x_b, IN, Wihb_b, IN, bih_b, gib, G, nullptr, 0, nullptr, 0, G, IN);

  // --- sequential recurrences (both directions concurrent) ---
  gru2_kernel<<<2, 768, 0, stream>>>(gif, gib, Whh_f, Whh_b, bhh_f, bhh_b, out_b, SEQ, TGT);
  fill_uid<<<(TGT * 64 + 255) / 256, 256, 0, stream>>>(embU, uid, out_b, TGT);

  // --- attention ---
  gemm_nt<0, 0, 1, 0, 0, 0><<<dim3(TGT / 128, NH / 128), 128, 0, stream>>>(
      out_b, KO, hist_b, H2, nullptr, scores, NH, nullptr, 0, nullptr, 0, NH, H2);
  softmax_rows<<<TGT, 256, 0, stream>>>(scores, attn_b, NH);
  gemm_nt<0, 0, 0, 1, 0, 0><<<dim3(TGT / 128, H2 / 128), 128, 0, stream>>>(
      attn_b, NH, histT_b, NH, nullptr, nullptr, 0, out_b + H2, KO, nullptr, 0, H2, NH);

  // --- classifier: out(1024x1088) @ Wf.T (fp32 B converted in-flight) ---
  gemm_nt<1, 0, 1, 0, 0, 1><<<dim3(TGT / 128, (LOC + 127) / 128), 128, 0, stream>>>(
      out_b, KO, Wf, KO, bf_bias, (float*)d_out, LOC, nullptr, 0, nullptr, 0, LOC, KO);

  // --- log_softmax ---
  row_lse<<<TGT, 512, 0, stream>>>((float*)d_out, lse, LOC);
  const long total = (long)TGT * LOC;
  if ((LOC & 3) == 0) {
    const long n4 = total >> 2;
    log_sub_vec<<<(n4 + 255) / 256, 256, 0, stream>>>((float*)d_out, lse, LOC, n4);
  } else {
    log_sub_sc<<<(total + 255) / 256, 256, 0, stream>>>((float*)d_out, lse, LOC, total);
  }
}
